// GATv2Regressor_76330158784604
// MI455X (gfx1250) — matrix operand order repaired and checked by reading, not compiled, not run on hardware
//
#include <hip/hip_runtime.h>
#include <hip/hip_bf16.h>

// ---------------------------------------------------------------------------
// GATv2 (2 layers) + GlobalAttention pool + MLP head, CDNA5 / gfx1250.
//  - Dense node transforms: f16 WMMA (v_wmma_f32_16x16x32_f16), weights
//    packed transposed-f16 and staged in LDS.
//  - Edge passes: wave-per-(edge,head), lane-per-channel (C==32==wave32).
//  - Segment softmax via order-preserving-uint atomicMax + f32 atomicAdd.
// ---------------------------------------------------------------------------

typedef __attribute__((ext_vector_type(16))) _Float16 v16h;
typedef __attribute__((ext_vector_type(8)))  float    v8f;

#define NEG_SLOPE 0.2f
#define NEG_INF_KEY 0x007FFFFFu   // fkey(-inf)

__device__ __forceinline__ unsigned fkey(float f) {
  unsigned u = __float_as_uint(f);
  return (u & 0x80000000u) ? ~u : (u | 0x80000000u);
}
__device__ __forceinline__ float fdec(unsigned k) {
  unsigned u = (k & 0x80000000u) ? (k ^ 0x80000000u) : ~k;
  return __uint_as_float(u);
}

// ---------------- prep kernels ----------------

// Wt[(n+colOff)*ldt + k] = (f16) W[k*Nout + n]   (transpose + convert)
__global__ void pack_wt_kernel(const float* __restrict__ W, _Float16* __restrict__ Wt,
                               int K, int Nout, int ldt, int colOff) {
  int idx = blockIdx.x * blockDim.x + threadIdx.x;
  if (idx >= K * Nout) return;
  int k = idx / Nout, n = idx % Nout;
  Wt[(size_t)(n + colOff) * ldt + k] = (_Float16)W[idx];
}

__global__ void cvt_f16_kernel(const float* __restrict__ src, _Float16* __restrict__ dst, int n) {
  int idx = blockIdx.x * blockDim.x + threadIdx.x;
  if (idx < n) dst[idx] = (_Float16)src[idx];
}

__global__ void fill_u32_kernel(unsigned* __restrict__ p, unsigned v, int n) {
  int idx = blockIdx.x * blockDim.x + threadIdx.x;
  if (idx < n) p[idx] = v;
}

// ---------------- WMMA GEMM: C[M,Nout] = A_f16[M,K] @ Bt_f16[Nout,K]^T -----
// One 16x16 output tile per wave; weights staged in LDS; K,Nout mult of 32/16,
// M mult of 16 (N=100000 = 6250*16 here).
__global__ __launch_bounds__(256)
void gemm_f16_wmma_kernel(const _Float16* __restrict__ A,
                          const _Float16* __restrict__ Bt,
                          float* __restrict__ C,
                          int M, int K, int Nout) {
  extern __shared__ _Float16 sW[];
  const int nW = K * Nout;
  for (int i = threadIdx.x; i < nW; i += blockDim.x) sW[i] = Bt[i];
  __syncthreads();

  const int lane = threadIdx.x & 31;
  const int wid  = blockIdx.x * (blockDim.x >> 5) + (threadIdx.x >> 5);
  const int mTiles = M >> 4;
  const int nTiles = Nout >> 4;
  if (wid >= mTiles * nTiles) return;           // wave-uniform: EXEC stays full
  const int mBase = (wid % mTiles) << 4;
  const int nBase = (wid / mTiles) << 4;

  const int g = lane >> 4;                      // half-wave group
  const int l = lane & 15;

  v8f acc = {0.f, 0.f, 0.f, 0.f, 0.f, 0.f, 0.f, 0.f};
  const _Float16* aRow = A  + (size_t)(mBase + l) * K;   // A row m = mBase+l
  const _Float16* bRow = sW + (size_t)(nBase + l) * K;   // Bt row n = nBase+l

  for (int k0 = 0; k0 < K; k0 += 32) {
    v16h a, b;
#pragma unroll
    for (int j = 0; j < 8; ++j) {
      // A 16x32 f16 layout (ISA 7.12.2): reg j, group g -> K pair
      int ka = k0 + ((j < 4) ? 0 : 16) + 8 * g + 2 * (j & 3);
      a[2 * j]     = aRow[ka];
      a[2 * j + 1] = aRow[ka + 1];
      // B 32x16 f16 layout: same K order as A (reg j, group g -> K pair)
      int kb = k0 + ((j < 4) ? 0 : 16) + 8 * g + 2 * (j & 3);
      b[2 * j]     = bRow[kb];
      b[2 * j + 1] = bRow[kb + 1];
    }
    acc = __builtin_amdgcn_wmma_f32_16x16x32_f16(false, a, false, b,
                                                 (short)0, acc, false, false);
  }

  // C/D layout: reg r -> row mBase + 8*g + r, col nBase + l
  float* cBase = C + (size_t)(mBase + 8 * g) * Nout + nBase + l;
#pragma unroll
  for (int r = 0; r < 8; ++r) cBase[(size_t)r * Nout] = acc[r];
}

// ---------------- edge pass 1: score + segment max ----------------
// wave per (edge,head); lane = channel (C==32)
__global__ __launch_bounds__(256)
void edge_score_kernel(const float* __restrict__ xl, const float* __restrict__ xr,
                       int ld, const int* __restrict__ src, const int* __restrict__ dst,
                       const float* __restrict__ att, float* __restrict__ score,
                       unsigned* __restrict__ nmax, int Eh, int Hh) {
  int wid  = (blockIdx.x * blockDim.x + threadIdx.x) >> 5;
  int lane = threadIdx.x & 31;
  if (wid >= Eh) return;
  int e = wid / Hh, h = wid - e * Hh;
  int s = src[e], d = dst[e];
  float v = xl[(size_t)s * ld + h * 32 + lane] + xr[(size_t)d * ld + h * 32 + lane];
  v = v > 0.f ? v : NEG_SLOPE * v;              // leaky_relu
  float p = v * att[h * 32 + lane];
#pragma unroll
  for (int m = 16; m; m >>= 1) p += __shfl_xor(p, m, 32);
  if (lane == 0) {
    score[wid] = p;
    atomicMax(nmax + (size_t)d * Hh + h, fkey(p));
  }
}

// ---------------- edge pass 2: exp + segment denom ----------------
__global__ void edge_exp_kernel(float* __restrict__ score, const int* __restrict__ dst,
                                const unsigned* __restrict__ nmax, float* __restrict__ den,
                                int Eh, int Hh) {
  int idx = blockIdx.x * blockDim.x + threadIdx.x;
  if (idx >= Eh) return;
  int e = idx / Hh, h = idx - e * Hh;
  int d = dst[e];
  float m = fdec(nmax[(size_t)d * Hh + h]);
  if (!(m > -3.0e38f)) m = 0.f;                 // nodes with no in-edges
  float ex = __expf(score[idx] - m);
  score[idx] = ex;                              // overwrite score with exp
  atomicAdd(den + (size_t)d * Hh + h, ex);
}

// ---------------- edge pass 3: alpha-weighted scatter-add ----------------
__global__ __launch_bounds__(256)
void edge_agg_kernel(const float* __restrict__ xl, int ld,
                     const int* __restrict__ src, const int* __restrict__ dst,
                     const float* __restrict__ ex, const float* __restrict__ den,
                     float* __restrict__ agg, int Eh, int Hh) {
  int wid  = (blockIdx.x * blockDim.x + threadIdx.x) >> 5;
  int lane = threadIdx.x & 31;
  if (wid >= Eh) return;
  int e = wid / Hh, h = wid - e * Hh;
  int s = src[e], d = dst[e];
  float alpha = ex[wid] / den[(size_t)d * Hh + h];
  float v = alpha * xl[(size_t)s * ld + h * 32 + lane];
  atomicAdd(agg + (size_t)d * (Hh * 32) + h * 32 + lane, v);
}

// ---------------- bias + relu epilogues ----------------
__global__ void node_relu_f16_kernel(const float* __restrict__ agg, const float* __restrict__ bias,
                                     _Float16* __restrict__ out, int n, int width) {
  int idx = blockIdx.x * blockDim.x + threadIdx.x;
  if (idx >= n) return;
  float v = agg[idx] + bias[idx % width];
  out[idx] = (_Float16)(v > 0.f ? v : 0.f);
}

__global__ void node_relu_f32_kernel(const float* __restrict__ agg, const float* __restrict__ bias,
                                     float* __restrict__ out, int n, int width) {
  int idx = blockIdx.x * blockDim.x + threadIdx.x;
  if (idx >= n) return;
  float v = agg[idx] + bias[idx % width];
  out[idx] = v > 0.f ? v : 0.f;
}

// ---------------- gate MLP: wave per node, lane = hidden channel ----------
__global__ __launch_bounds__(256)
void gate_kernel(const float* __restrict__ h2, const float* __restrict__ g1w,
                 const float* __restrict__ g1b, const float* __restrict__ g2w,
                 const float* __restrict__ g2b, const int* __restrict__ batch,
                 float* __restrict__ gate, unsigned* __restrict__ gmax, int n) {
  int wid  = (blockIdx.x * blockDim.x + threadIdx.x) >> 5;
  int lane = threadIdx.x & 31;
  if (wid >= n) return;
  const float* hp = h2 + (size_t)wid * 32;
  float t = g1b[lane];
#pragma unroll
  for (int k = 0; k < 32; ++k) t = fmaf(hp[k], g1w[k * 32 + lane], t);
  t = t > 0.f ? t : 0.f;
  float p = t * g2w[lane];
#pragma unroll
  for (int m = 16; m; m >>= 1) p += __shfl_xor(p, m, 32);
  if (lane == 0) {
    float g = p + g2b[0];
    gate[wid] = g;
    atomicMax(gmax + batch[wid], fkey(g));
  }
}

// ---------------- pooling softmax ----------------
__global__ void pool_exp_kernel(const float* __restrict__ gate, const int* __restrict__ batch,
                                const unsigned* __restrict__ gmax, float* __restrict__ gden,
                                float* __restrict__ gexp, int n) {
  int idx = blockIdx.x * blockDim.x + threadIdx.x;
  if (idx >= n) return;
  int b = batch[idx];
  float m = fdec(gmax[b]);
  if (!(m > -3.0e38f)) m = 0.f;
  float ex = __expf(gate[idx] - m);
  gexp[idx] = ex;
  atomicAdd(gden + b, ex);
}

__global__ __launch_bounds__(256)
void pool_agg_kernel(const float* __restrict__ h2, const float* __restrict__ gexp,
                     const float* __restrict__ gden, const int* __restrict__ batch,
                     float* __restrict__ pooled, int n) {
  int wid  = (blockIdx.x * blockDim.x + threadIdx.x) >> 5;
  int lane = threadIdx.x & 31;
  if (wid >= n) return;
  int b = batch[wid];
  float w = gexp[wid] / gden[b];
  atomicAdd(pooled + (size_t)b * 32 + lane, w * h2[(size_t)wid * 32 + lane]);
}

// ---------------- head MLP: thread per graph ----------------
__global__ void head_kernel(const float* __restrict__ pooled, const float* __restrict__ lin1w,
                            const float* __restrict__ lin1b, const float* __restrict__ lin2w,
                            const float* __restrict__ lin2b, float* __restrict__ out, int G) {
  int g = blockIdx.x * blockDim.x + threadIdx.x;
  if (g >= G) return;
  const float* pp = pooled + (size_t)g * 32;
  float acc = lin2b[0];
  for (int c = 0; c < 32; ++c) {
    float z = lin1b[c];
#pragma unroll
    for (int k = 0; k < 32; ++k) z = fmaf(pp[k], lin1w[k * 32 + c], z);
    z = z > 0.f ? z : 0.f;
    acc = fmaf(z, lin2w[c], acc);
  }
  out[g] = acc;
}

// ---------------------------------------------------------------------------

extern "C" void kernel_launch(void* const* d_in, const int* in_sizes, int n_in,
                              void* d_out, int out_size, void* d_ws, size_t ws_size,
                              hipStream_t stream) {
  (void)n_in; (void)ws_size;
  const int IN = 128, C = 32, H = 2;
  const int n_nodes  = in_sizes[0] / IN;     // 100000 (multiple of 16)
  const int n_edges  = in_sizes[1] / 2;      // 1600000
  const int n_graphs = out_size;             // 64

  const float* x     = (const float*)d_in[0];
  const int*   src   = (const int*)d_in[1];
  const int*   dst   = src + n_edges;
  const int*   batch = (const int*)d_in[2];
  const float* Wl1 = (const float*)d_in[3];
  const float* Wr1 = (const float*)d_in[4];
  const float* att1 = (const float*)d_in[5];
  const float* b1   = (const float*)d_in[6];
  const float* Wl2 = (const float*)d_in[7];
  const float* Wr2 = (const float*)d_in[8];
  const float* att2 = (const float*)d_in[9];
  const float* b2   = (const float*)d_in[10];
  const float* g1w = (const float*)d_in[11];
  const float* g1b = (const float*)d_in[12];
  const float* g2w = (const float*)d_in[13];
  const float* g2b = (const float*)d_in[14];
  const float* lin1w = (const float*)d_in[15];
  const float* lin1b = (const float*)d_in[16];
  const float* lin2w = (const float*)d_in[17];
  const float* lin2b = (const float*)d_in[18];

  // ---- workspace layout ----
  char* ws = (char*)d_ws;
  size_t off = 0;
  auto take = [&](size_t bytes) -> char* {
    char* p = ws + off;
    off += (bytes + 255) & ~(size_t)255;
    return p;
  };
  _Float16* x16   = (_Float16*)take((size_t)n_nodes * IN * 2);
  _Float16* W1t   = (_Float16*)take((size_t)128 * 128 * 2);   // [Wl1|Wr1]^T f16
  _Float16* W2t   = (_Float16*)take((size_t)64 * 64 * 2);     // [Wl2|Wr2]^T f16
  float*    xlr1  = (float*)take((size_t)n_nodes * 128 * 4);  // [xl1 | xr1]
  _Float16* h1_16 = (_Float16*)take((size_t)n_nodes * 64 * 2);
  float*    xlr2  = (float*)take((size_t)n_nodes * 64 * 4);   // [xl2 | xr2]
  float*    score1 = (float*)take((size_t)n_edges * H * 4);
  float*    score2 = (float*)take((size_t)n_edges * 4);
  unsigned* nmax1  = (unsigned*)take((size_t)n_nodes * H * 4);
  float*    nden1  = (float*)take((size_t)n_nodes * H * 4);
  float*    agg1   = (float*)take((size_t)n_nodes * 64 * 4);
  unsigned* nmax2  = (unsigned*)take((size_t)n_nodes * 4);
  float*    nden2  = (float*)take((size_t)n_nodes * 4);
  float*    agg2   = (float*)take((size_t)n_nodes * 32 * 4);
  float*    h2     = (float*)take((size_t)n_nodes * 32 * 4);
  float*    gate   = (float*)take((size_t)n_nodes * 4);
  float*    gexp   = (float*)take((size_t)n_nodes * 4);
  unsigned* gmax   = (unsigned*)take((size_t)n_graphs * 4);
  float*    gden   = (float*)take((size_t)n_graphs * 4);
  float*    pooled = (float*)take((size_t)n_graphs * 32 * 4);

  const int TB = 256;
  auto blocks = [](long long work, int tb) { return (unsigned)((work + tb - 1) / tb); };

  // ---- prep: pack weights transposed-f16, convert x ----
  pack_wt_kernel<<<blocks(128 * 64, TB), TB, 0, stream>>>(Wl1, W1t, 128, 64, 128, 0);
  pack_wt_kernel<<<blocks(128 * 64, TB), TB, 0, stream>>>(Wr1, W1t, 128, 64, 128, 64);
  pack_wt_kernel<<<blocks(64 * 32, TB), TB, 0, stream>>>(Wl2, W2t, 64, 32, 64, 0);
  pack_wt_kernel<<<blocks(64 * 32, TB), TB, 0, stream>>>(Wr2, W2t, 64, 32, 64, 32);
  cvt_f16_kernel<<<blocks((long long)n_nodes * IN, TB), TB, 0, stream>>>(x, x16, n_nodes * IN);

  // ---- init accumulators (fresh every call; deterministic) ----
  hipMemsetAsync(nden1, 0, (size_t)n_nodes * H * 4, stream);
  hipMemsetAsync(agg1, 0, (size_t)n_nodes * 64 * 4, stream);
  hipMemsetAsync(nden2, 0, (size_t)n_nodes * 4, stream);
  hipMemsetAsync(agg2, 0, (size_t)n_nodes * 32 * 4, stream);
  hipMemsetAsync(gden, 0, (size_t)n_graphs * 4, stream);
  hipMemsetAsync(pooled, 0, (size_t)n_graphs * 32 * 4, stream);
  fill_u32_kernel<<<blocks(n_nodes * H, TB), TB, 0, stream>>>(nmax1, NEG_INF_KEY, n_nodes * H);
  fill_u32_kernel<<<blocks(n_nodes, TB), TB, 0, stream>>>(nmax2, NEG_INF_KEY, n_nodes);
  fill_u32_kernel<<<blocks(n_graphs, TB), TB, 0, stream>>>(gmax, NEG_INF_KEY, n_graphs);

  // ---- layer 1: fused [xl|xr] = x @ [Wl1|Wr1]  (WMMA) ----
  {
    int tiles = (n_nodes / 16) * (128 / 16);
    gemm_f16_wmma_kernel<<<blocks((long long)tiles * 32, TB), TB,
                           128 * 128 * sizeof(_Float16), stream>>>(
        x16, W1t, xlr1, n_nodes, 128, 128);
  }
  // ---- layer 1 edge softmax + aggregate ----
  {
    int Eh = n_edges * H;
    edge_score_kernel<<<blocks((long long)Eh * 32, TB), TB, 0, stream>>>(
        xlr1, xlr1 + 64, 128, src, dst, att1, score1, nmax1, Eh, H);
    edge_exp_kernel<<<blocks(Eh, TB), TB, 0, stream>>>(score1, dst, nmax1, nden1, Eh, H);
    edge_agg_kernel<<<blocks((long long)Eh * 32, TB), TB, 0, stream>>>(
        xlr1, 128, src, dst, score1, nden1, agg1, Eh, H);
  }
  node_relu_f16_kernel<<<blocks((long long)n_nodes * 64, TB), TB, 0, stream>>>(
      agg1, b1, h1_16, n_nodes * 64, 64);

  // ---- layer 2: fused [xl|xr] = h1 @ [Wl2|Wr2]  (WMMA) ----
  {
    int tiles = (n_nodes / 16) * (64 / 16);
    gemm_f16_wmma_kernel<<<blocks((long long)tiles * 32, TB), TB,
                           64 * 64 * sizeof(_Float16), stream>>>(
        h1_16, W2t, xlr2, n_nodes, 64, 64);
  }
  // ---- layer 2 edge softmax + aggregate (H=1) ----
  edge_score_kernel<<<blocks((long long)n_edges * 32, TB), TB, 0, stream>>>(
      xlr2, xlr2 + 32, 64, src, dst, att2, score2, nmax2, n_edges, 1);
  edge_exp_kernel<<<blocks(n_edges, TB), TB, 0, stream>>>(score2, dst, nmax2, nden2, n_edges, 1);
  edge_agg_kernel<<<blocks((long long)n_edges * 32, TB), TB, 0, stream>>>(
      xlr2, 64, src, dst, score2, nden2, agg2, n_edges, 1);
  node_relu_f32_kernel<<<blocks((long long)n_nodes * 32, TB), TB, 0, stream>>>(
      agg2, b2, h2, n_nodes * 32, 32);

  // ---- global attention pooling ----
  gate_kernel<<<blocks((long long)n_nodes * 32, TB), TB, 0, stream>>>(
      h2, g1w, g1b, g2w, g2b, batch, gate, gmax, n_nodes);
  pool_exp_kernel<<<blocks(n_nodes, TB), TB, 0, stream>>>(gate, batch, gmax, gden, gexp, n_nodes);
  pool_agg_kernel<<<blocks((long long)n_nodes * 32, TB), TB, 0, stream>>>(
      h2, gexp, gden, batch, pooled, n_nodes);

  // ---- head MLP ----
  head_kernel<<<blocks(n_graphs, 64), 64, 0, stream>>>(
      pooled, lin1w, lin1b, lin2w, lin2b, (float*)d_out, n_graphs);
}
